// Bottleneck_37503654429269
// MI455X (gfx1250) — 8-bit matrix operand order repaired and checked by reading, compiled, not run on hardware
//
#include <hip/hip_runtime.h>

typedef int v8i __attribute__((ext_vector_type(8)));
typedef int vi4 __attribute__((vector_size(16)));
typedef __attribute__((address_space(1))) vi4 as1_vi4;
typedef __attribute__((address_space(3))) vi4 as3_vi4;

#define BATCH 32
#define PIX   196
#define CIN   1024
#define CW    256
#define COUT  1024
#define K2    2304

#if defined(__has_builtin)
#if __has_builtin(__builtin_amdgcn_global_load_async_to_lds_b128)
#define HAVE_ASYNC_LDS 1
#endif
#endif

#define OFF_QW1  (8192)
#define OFF_QW2  (OFF_QW1 + 3*256*1024)
#define OFF_QW3  (OFF_QW2 + 3*256*2304)
#define OFF_ACT0 (OFF_QW3 + 3*1024*256)
#define OFF_H1   (OFF_ACT0 + 32*196*1024)
#define OFF_ZROW (OFF_H1  + 32*196*256)
#define OFF_H2   (OFF_ZROW + 1024)

__device__ __forceinline__ float lvm1_of(int e) {
  return (e == 0) ? 3.f : (e == 1) ? 15.f : 255.f;
}
__device__ __forceinline__ float nlev_of(int e) {
  return (e == 0) ? 1.f : (e == 1) ? 7.f : 127.f;
}

__global__ void absmax_k(const float* __restrict__ w, int n, float* __restrict__ out) {
  __shared__ float red[256];
  float m = 0.f;
  for (int i = blockIdx.x * 256 + threadIdx.x; i < n; i += gridDim.x * 256)
    m = fmaxf(m, fabsf(w[i]));
  red[threadIdx.x] = m;
  __syncthreads();
  for (int s = 128; s > 0; s >>= 1) {
    if (threadIdx.x < s) red[threadIdx.x] = fmaxf(red[threadIdx.x], red[threadIdx.x + s]);
    __syncthreads();
  }
  if (threadIdx.x == 0)
    atomicMax((unsigned int*)out, __float_as_uint(red[0]));
}

__global__ void prep_k(const float* __restrict__ b1g, const float* __restrict__ b1b,
                       const float* __restrict__ b1m, const float* __restrict__ b1v,
                       const float* __restrict__ b2g, const float* __restrict__ b2b,
                       const float* __restrict__ b2m, const float* __restrict__ b2v,
                       float* __restrict__ fbuf) {
  int t = threadIdx.x;
  float inv1 = b1g[t] * rsqrtf(b1v[t] + 1e-5f);
  fbuf[32 + t]  = inv1;
  fbuf[288 + t] = b1b[t] - b1m[t] * inv1;
  float inv2 = b2g[t] * rsqrtf(b2v[t] + 1e-5f);
  fbuf[544 + t] = inv2;
  fbuf[800 + t] = b2b[t] - b2m[t] * inv2;
  if (t < 9) {
    int e = t / 3, ti = t % 3;
    float s = fbuf[ti] + 1e-12f;
    fbuf[16 + t] = (s / nlev_of(e)) / lvm1_of(e);
  }
}

__global__ void quantw_k(const float* __restrict__ w, signed char* __restrict__ q,
                         int n, const float* __restrict__ fbuf, int ti) {
  int i = blockIdx.x * 256 + threadIdx.x;
  if (i >= 3 * n) return;
  int e = i / n, j = i % n;
  float s = fbuf[ti] + 1e-12f;
  q[i] = (signed char)__float2int_rn(w[j] / s * nlev_of(e));
}

__global__ void quantw2_k(const float* __restrict__ w, signed char* __restrict__ q,
                          const float* __restrict__ fbuf) {
  const int n = 256 * 2304;
  int idx = blockIdx.x * 256 + threadIdx.x;
  if (idx >= 3 * n) return;
  int e = idx / n, r = idx % n;
  int o = r / 2304, k = r % 2304;
  int tap = k >> 8, i = k & 255;
  float s = fbuf[1] + 1e-12f;
  q[idx] = (signed char)__float2int_rn(w[(o * 256 + i) * 9 + tap] / s * nlev_of(e));
}

__global__ void qact0_k(const float* __restrict__ x, unsigned char* __restrict__ act,
                        const int* __restrict__ mask) {
  int idx = blockIdx.x * 256 + threadIdx.x;
  if (idx >= BATCH * PIX * CIN) return;
  int c = idx & (CIN - 1);
  int bp = idx >> 10;
  int p = bp % PIX, b = bp / PIX;
  float lvm1 = lvm1_of(mask[b]);
  float v = x[(b * CIN + c) * PIX + p];
  v = fminf(fmaxf(v, 0.f), 1.f);
  act[idx] = (unsigned char)__float2int_rn(v * lvm1);
}

template <int K, int M, int STAGE>
__global__ __launch_bounds__(128) void gemm_lds(
    const signed char* __restrict__ qw, const unsigned char* __restrict__ act,
    const unsigned char* __restrict__ zrow, void* __restrict__ outp,
    const int* __restrict__ mask, const float* __restrict__ fbuf) {
  __shared__ __align__(16) unsigned char sAct[16 * K];
  int b = blockIdx.z;
  int e = mask[b];
  int p0 = blockIdx.y * 16;
  int tid = threadIdx.x;

  {
    const int vpr = K / 16;
#if defined(HAVE_ASYNC_LDS)
#pragma unroll
    for (int i0 = 0; i0 < 16 * vpr; i0 += 128) {
      int i = i0 + tid;
      int r = i / vpr, col = i - r * vpr, p = p0 + r;
      const unsigned char* gsrc =
          (p < PIX) ? (act + b * (PIX * K) + p * K + col * 16) : (zrow + col * 16);
      __builtin_amdgcn_global_load_async_to_lds_b128(
          (as1_vi4*)(unsigned char*)gsrc,
          (as3_vi4*)(sAct + i * 16), 0, 0);
    }
    asm volatile("s_wait_asynccnt 0" ::: "memory");
#else
    const uint4* src = (const uint4*)(act + b * (PIX * K));
    uint4* dst = (uint4*)sAct;
#pragma unroll
    for (int i0 = 0; i0 < 16 * vpr; i0 += 128) {
      int i = i0 + tid;
      int r = i / vpr, p = p0 + r;
      uint4 v = {0u, 0u, 0u, 0u};
      if (p < PIX) v = src[p * vpr + (i - r * vpr)];
      dst[i] = v;
    }
#endif
  }
  __syncthreads();

  int wave = tid >> 5, lane = tid & 31;
  int ln = lane & 15, hi = lane >> 4;
  int mtile = blockIdx.x * 4 + wave;
  const signed char* wbase = qw + (e * M + mtile * 16 + ln) * K;
  const unsigned char* brow = sAct + ln * K;

  v8i acc = {0, 0, 0, 0, 0, 0, 0, 0};
#pragma unroll 4
  for (int kt = 0; kt < K / 64; ++kt) {
    int k0 = kt * 64;
    v8i A, Bv;
#pragma unroll
    for (int v = 0; v < 8; ++v) {
      int ak = k0 + (hi << 4) + ((v & 3) << 2) + ((v >> 2) << 5);
      A[v] = *(const int*)(wbase + ak);
      int bk = k0 + (hi << 4) + ((v & 3) << 2) + ((v >> 2) << 5);
      Bv[v] = *(const int*)(brow + bk);
    }
    acc = __builtin_amdgcn_wmma_i32_16x16x64_iu8(true, A, false, Bv, acc, false, false);
  }

  float cs = fbuf[16 + e * 3 + ((STAGE == 1) ? 0 : 2)];
  int p = p0 + ln;
  if (p < PIX) {
    if (STAGE == 1) {
      float lvm1 = lvm1_of(e);
      unsigned char* o = (unsigned char*)outp + (b * PIX + p) * M;
#pragma unroll
      for (int v = 0; v < 8; ++v) {
        int m = mtile * 16 + hi * 8 + v;
        float f = (float)acc[v] * cs * fbuf[32 + m] + fbuf[288 + m];
        f = fminf(fmaxf(f, 0.f), 1.f);
        o[m] = (unsigned char)__float2int_rn(f * lvm1);
      }
    } else {
      float* o = (float*)outp;
#pragma unroll
      for (int v = 0; v < 8; ++v) {
        int m = mtile * 16 + hi * 8 + v;
        o[(b * M + m) * PIX + p] = (float)acc[v] * cs;
      }
    }
  }
}

__global__ __launch_bounds__(128) void gemm_s2(
    const signed char* __restrict__ qw, const unsigned char* __restrict__ h1,
    const unsigned char* __restrict__ zrow, unsigned char* __restrict__ h2,
    const int* __restrict__ mask, const float* __restrict__ fbuf) {
  int b = blockIdx.z;
  int e = mask[b];
  int lane = threadIdx.x & 31, wave = threadIdx.x >> 5;
  int ln = lane & 15, hi = lane >> 4;
  int mtile = blockIdx.x * 4 + wave;
  const signed char* wbase = qw + (e * CW + mtile * 16 + ln) * K2;
  int p = blockIdx.y * 16 + ln;
  int h = p / 14, w = p % 14;
  const unsigned char* abase = h1 + b * (PIX * CW);
  bool pv = (p < PIX);

  v8i acc = {0, 0, 0, 0, 0, 0, 0, 0};
#pragma unroll
  for (int tap = 0; tap < 9; ++tap) {
    const int dy = tap / 3 - 1, dx = tap % 3 - 1;
    int hh = h + dy, ww = w + dx;
    bool valid = pv && ((unsigned)hh < 14u) && ((unsigned)ww < 14u);
    const unsigned char* arow = valid ? (abase + (hh * 14 + ww) * CW) : zrow;
#pragma unroll
    for (int ct = 0; ct < 4; ++ct) {
      const int kt = tap * 4 + ct;
      v8i A, Bv;
#pragma unroll
      for (int v = 0; v < 8; ++v) {
        const int ak = kt * 64 + ((v & 3) << 2) + ((v >> 2) << 5);
        A[v] = *(const int*)(wbase + ak + (hi << 4));
        const int bk = ct * 64 + ((v & 3) << 2) + ((v >> 2) << 5);
        Bv[v] = *(const int*)(arow + bk + (hi << 4));
      }
      acc = __builtin_amdgcn_wmma_i32_16x16x64_iu8(true, A, false, Bv, acc, false, false);
    }
  }

  float cs = fbuf[16 + e * 3 + 1];
  if (pv) {
    float lvm1 = lvm1_of(e);
    unsigned char* o = h2 + (b * PIX + p) * CW;
#pragma unroll
    for (int v = 0; v < 8; ++v) {
      int m = mtile * 16 + hi * 8 + v;
      float f = (float)acc[v] * cs * fbuf[544 + m] + fbuf[800 + m];
      f = fminf(fmaxf(f, 0.f), 1.f);
      o[m] = (unsigned char)__float2int_rn(f * lvm1);
    }
  }
}

__global__ void gnstats_k(const float* __restrict__ out, float* __restrict__ fbuf) {
  int b = blockIdx.x >> 2, g = blockIdx.x & 3;
  const float* base = out + (b * COUT + g * 256) * PIX;
  float s = 0.f, ss = 0.f;
  for (int i = threadIdx.x; i < 256 * PIX; i += 256) {
    float v = base[i];
    s += v;
    ss += v * v;
  }
  __shared__ float rs[256], rq[256];
  rs[threadIdx.x] = s;
  rq[threadIdx.x] = ss;
  __syncthreads();
  for (int st = 128; st > 0; st >>= 1) {
    if (threadIdx.x < st) {
      rs[threadIdx.x] += rs[threadIdx.x + st];
      rq[threadIdx.x] += rq[threadIdx.x + st];
    }
    __syncthreads();
  }
  if (threadIdx.x == 0) {
    const float inv_n = 1.f / (256.f * (float)PIX);
    float mu = rs[0] * inv_n;
    float var = rq[0] * inv_n - mu * mu;
    fbuf[1056 + blockIdx.x] = mu;
    fbuf[1184 + blockIdx.x] = rsqrtf(var + 1e-5f);
  }
}

__global__ void gnapply_k(float* __restrict__ out, const float* __restrict__ x,
                          const float* __restrict__ gng, const float* __restrict__ gnb,
                          const float* __restrict__ fbuf) {
  int idx = blockIdx.x * 256 + threadIdx.x;
  if (idx >= BATCH * COUT * PIX) return;
  int c = (idx / PIX) & (COUT - 1);
  int b = idx / (COUT * PIX);
  int g = c >> 8;
  float mu = fbuf[1056 + b * 4 + g];
  float is = fbuf[1184 + b * 4 + g];
  float v = (out[idx] - mu) * is * gng[c] + gnb[c] + x[idx];
  out[idx] = fmaxf(v, 0.f);
}

extern "C" void kernel_launch(void* const* d_in, const int* in_sizes, int n_in,
                              void* d_out, int out_size, void* d_ws, size_t ws_size,
                              hipStream_t stream) {
  (void)in_sizes; (void)n_in; (void)out_size; (void)ws_size;
  const float* x    = (const float*)d_in[0];
  const int*   mask = (const int*)d_in[1];
  const float* w1   = (const float*)d_in[2];
  const float* w2   = (const float*)d_in[3];
  const float* w3   = (const float*)d_in[4];
  const float* b1g  = (const float*)d_in[5];
  const float* b1b  = (const float*)d_in[6];
  const float* b1m  = (const float*)d_in[7];
  const float* b1v  = (const float*)d_in[8];
  const float* b2g  = (const float*)d_in[9];
  const float* b2b  = (const float*)d_in[10];
  const float* b2m  = (const float*)d_in[11];
  const float* b2v  = (const float*)d_in[12];
  const float* gng  = (const float*)d_in[13];
  const float* gnb  = (const float*)d_in[14];
  float* out = (float*)d_out;

  char* ws = (char*)d_ws;
  float* fbuf = (float*)ws;
  signed char*   qw1  = (signed char*)(ws + OFF_QW1);
  signed char*   qw2  = (signed char*)(ws + OFF_QW2);
  signed char*   qw3  = (signed char*)(ws + OFF_QW3);
  unsigned char* act0 = (unsigned char*)(ws + OFF_ACT0);
  unsigned char* h1   = (unsigned char*)(ws + OFF_H1);
  unsigned char* zrow = (unsigned char*)(ws + OFF_ZROW);
  unsigned char* h2   = (unsigned char*)(ws + OFF_H2);

  (void)hipMemsetAsync(fbuf, 0, 16, stream);
  (void)hipMemsetAsync(zrow, 0, 1024, stream);
  absmax_k<<<64, 256, 0, stream>>>(w1, CW * CIN, fbuf + 0);
  absmax_k<<<64, 256, 0, stream>>>(w2, CW * CW * 9, fbuf + 1);
  absmax_k<<<64, 256, 0, stream>>>(w3, COUT * CW, fbuf + 2);
  prep_k<<<1, 256, 0, stream>>>(b1g, b1b, b1m, b1v, b2g, b2b, b2m, b2v, fbuf);

  quantw_k<<<(3 * CW * CIN + 255) / 256, 256, 0, stream>>>(w1, qw1, CW * CIN, fbuf, 0);
  quantw2_k<<<(3 * CW * K2 + 255) / 256, 256, 0, stream>>>(w2, qw2, fbuf);
  quantw_k<<<(3 * COUT * CW + 255) / 256, 256, 0, stream>>>(w3, qw3, COUT * CW, fbuf, 2);

  qact0_k<<<(BATCH * PIX * CIN + 255) / 256, 256, 0, stream>>>(x, act0, mask);

  gemm_lds<CIN, CW, 1><<<dim3(4, 13, BATCH), 128, 0, stream>>>(qw1, act0, zrow, (void*)h1, mask, fbuf);
  gemm_s2<<<dim3(4, 13, BATCH), 128, 0, stream>>>(qw2, h1, zrow, h2, mask, fbuf);
  gemm_lds<CW, COUT, 3><<<dim3(16, 13, BATCH), 128, 0, stream>>>(qw3, h2, zrow, (void*)out, mask, fbuf);

  gnstats_k<<<BATCH * 4, 256, 0, stream>>>(out, fbuf);
  gnapply_k<<<(BATCH * COUT * PIX + 255) / 256, 256, 0, stream>>>(out, x, gng, gnb, fbuf);
}
